// MLABase_41016937677183
// MI455X (gfx1250) — compile-verified, matrix operand order checked by reading, not run on hardware
//
#include <hip/hip_runtime.h>

// ---------------------------------------------------------------------------
// MLA projection front-end for MI455X (gfx1250, wave32, WMMA)
//   q_low  = x @ wq_a                (8192x2048 * 2048x1536)
//   q      = rmsnorm(q_low) @ wq_b   (8192x1536 * 1536x3072)
//   kvfull = x @ wkv_a               (8192x2048 * 2048x576)
//   + split / rotary epilogues
// GEMMs: bf16 v_wmma_f32_16x16x32_bf16, f32 accumulation.
// Wave tile 64x64 (4x4 WMMA), block tile 128x256 (8 waves, 2M x 4N).
// K,N are template constants -> tile offsets fold into load/store immediates.
// All GEMM traffic goes through address_space(1) pointers -> global_load/store
// (LOADcnt only; no FLAT/DScnt coupling in the hot loop).
// ---------------------------------------------------------------------------

typedef __attribute__((ext_vector_type(16))) __bf16          v16bf;
typedef __attribute__((ext_vector_type(8)))  float           v8f;
typedef __attribute__((ext_vector_type(8)))  unsigned short  v8u16;
typedef __attribute__((ext_vector_type(16))) unsigned short  v16u16;

typedef const unsigned short __attribute__((address_space(1)))* gu16cp;
typedef const v8u16          __attribute__((address_space(1)))* gv8cp;
typedef float                __attribute__((address_space(1)))* gf32p;

#define B_    2
#define S_    4096
#define DIM_  2048
#define NH_   16
#define QL_   1536
#define KVL_  512
#define QKN_  128
#define QKR_  64
#define QKH_  192
#define MTOT  (B_ * S_)          // 8192 rows for every GEMM

// ---------------- fp32 -> bf16 (round-to-nearest-even) ----------------------
__device__ __forceinline__ unsigned short f2bf(float f) {
    unsigned int u = __float_as_uint(f);
    u += 0x7FFFu + ((u >> 16) & 1u);
    return (unsigned short)(u >> 16);
}

__global__ void k_conv(const float* __restrict__ in, unsigned short* __restrict__ out, int n) {
    int i = blockIdx.x * blockDim.x + threadIdx.x;
    if (i < n) out[i] = f2bf(in[i]);
}

// in: [K,N] row-major fp32  ->  out: [N,K] row-major bf16 (transposed)
__global__ void k_conv_tr(const float* __restrict__ in, unsigned short* __restrict__ out,
                          int K, int N) {
    int i = blockIdx.x * blockDim.x + threadIdx.x;
    if (i < K * N) {
        int k = i / N;
        int n = i - k * N;
        out[(long)n * K + k] = f2bf(in[i]);
    }
}

// ---------------- WMMA fragment load (global AS) ----------------------------
// 16-bit A-matrix 16x32 layout (ISA 7.12.2):
//   lanes 0-15 : row M=lane,    K = k0+0..7  (bytes 0..15) and k0+16..23 (bytes 32..47)
//   lanes 16-31: row M=lane-16, K = k0+8..15            and k0+24..31
// p already includes the per-lane row base and the lane-half K offset.
__device__ __forceinline__ v16bf load_frag_g(gu16cp p) {
    gv8cp q = (gv8cp)p;
    v16u16 t;
    *((v8u16*)&t)     = q[0];   // bytes  0..15
    *((v8u16*)&t + 1) = q[2];   // bytes 32..47
    return __builtin_bit_cast(v16bf, t);
}

#define WMMA_BF16(a, b, c) \
    __builtin_amdgcn_wmma_f32_16x16x32_bf16(false, (a), false, (b), (short)0, (c), false, false)

// ---------------- bf16 WMMA GEMM:  C[M,N] = A[M,K] * Bt[N,K]^T --------------
// Compile-time K, N. Requires M % 128 == 0, N % 16 == 0, K % 32 == 0.
template <int K, int N>
__global__ __launch_bounds__(256)
void k_gemm_bf16(const unsigned short* __restrict__ A_,
                 const unsigned short* __restrict__ Bt_,
                 float* __restrict__ C_, int M) {
    gu16cp A  = (gu16cp)(unsigned long long)A_;
    gu16cp Bt = (gu16cp)(unsigned long long)Bt_;
    gf32p  C  = (gf32p)(unsigned long long)C_;

    const int lane  = threadIdx.x & 31;
    const int wave  = threadIdx.x >> 5;
    const int wm    = wave & 1;    // 0..1
    const int wn    = wave >> 1;   // 0..3
    const int bm    = blockIdx.y * 128 + wm * 64;
    const int bn    = blockIdx.x * 256 + wn * 64;
    const int rlane = lane & 15;
    const int koff  = (lane >> 4) << 3;   // lanes 16-31 start at K+8

    // One A address (tile offsets t*16*K are immediates); four B addresses
    // (per-tile row clamp for the N edge; clamped tiles are never stored).
    gu16cp ap = A + (long)(bm + rlane) * K + koff;
    gu16cp bp[4];
#pragma unroll
    for (int t = 0; t < 4; ++t) {
        int br = bn + t * 16 + rlane;
        if (br >= N) br = N - 1;           // tile-level clamp (N % 16 == 0)
        bp[t] = Bt + (long)br * K + koff;
    }

    v8f acc[4][4];
#pragma unroll
    for (int tm = 0; tm < 4; ++tm)
#pragma unroll
        for (int tn = 0; tn < 4; ++tn)
            acc[tm][tn] = (v8f){0.f, 0.f, 0.f, 0.f, 0.f, 0.f, 0.f, 0.f};

#pragma unroll 1
    for (int k0 = 0; k0 < K; k0 += 32) {
        v16bf a[4], b[4];
#pragma unroll
        for (int t = 0; t < 4; ++t) a[t] = load_frag_g(ap + t * (16 * K));
#pragma unroll
        for (int t = 0; t < 4; ++t) b[t] = load_frag_g(bp[t]);
        ap += 32;
#pragma unroll
        for (int t = 0; t < 4; ++t) bp[t] += 32;
#pragma unroll
        for (int tm = 0; tm < 4; ++tm)
#pragma unroll
            for (int tn = 0; tn < 4; ++tn)
                acc[tm][tn] = WMMA_BF16(a[tm], b[tn], acc[tm][tn]);
    }

    // C/D layout: VGPR i of lane l -> M = i + 8*(l>=16), N = l&15.
    // One base address; all tile/row offsets are immediates.
    const int half = lane >> 4;
    gf32p cp = C + (long)(bm + half * 8) * N + bn + rlane;
#pragma unroll
    for (int tn = 0; tn < 4; ++tn) {
        if (bn + tn * 16 < N) {            // uniform (scalar) guard
#pragma unroll
            for (int tm = 0; tm < 4; ++tm)
#pragma unroll
                for (int i = 0; i < 8; ++i)
                    cp[(long)(tm * 16 + i) * N + tn * 16] = acc[tm][tn][i];
        }
    }
}

// ---------------- RMSNorm over 1536-wide rows, emit bf16 --------------------
__global__ __launch_bounds__(256)
void k_rmsnorm(const float* __restrict__ in, const float* __restrict__ w,
               unsigned short* __restrict__ out) {
    __shared__ float red[256];
    const int row = blockIdx.x;
    const int t   = threadIdx.x;
    const float* r = in + (long)row * QL_;

    float v[6];
    float ss = 0.f;
#pragma unroll
    for (int j = 0; j < 6; ++j) {
        v[j] = r[t + j * 256];
        ss += v[j] * v[j];
    }
    red[t] = ss;
    __syncthreads();
#pragma unroll
    for (int s = 128; s > 0; s >>= 1) {
        if (t < s) red[t] += red[t + s];
        __syncthreads();
    }
    const float scale = rsqrtf(red[0] * (1.0f / (float)QL_) + 1e-6f);
#pragma unroll
    for (int j = 0; j < 6; ++j) {
        int c = t + j * 256;
        out[(long)row * QL_ + c] = f2bf(v[j] * scale * w[c]);
    }
}

// ---------------- Epilogues: split + rotary ---------------------------------
__global__ void k_copy_qnope(const float* __restrict__ qf, float* __restrict__ out, int n) {
    int i = blockIdx.x * blockDim.x + threadIdx.x;
    if (i < n) {
        int d = i & 127;
        int rest = i >> 7;
        int h = rest & 15;
        int bs = rest >> 4;
        out[i] = qf[(long)bs * (NH_ * QKH_) + h * QKH_ + d];
    }
}

__global__ void k_rot_qpe(const float* __restrict__ qf,
                          const float* __restrict__ fc, const float* __restrict__ fs,
                          float* __restrict__ out, int n) {   // n = B*S*16*32 pairs
    int i = blockIdx.x * blockDim.x + threadIdx.x;
    if (i < n) {
        int p = i & 31;
        int rest = i >> 5;
        int h = rest & 15;
        int bs = rest >> 4;
        int s = bs & (S_ - 1);
        float c  = fc[s * 32 + p];
        float sn = fs[s * 32 + p];
        const float* src = qf + (long)bs * (NH_ * QKH_) + h * QKH_ + QKN_ + 2 * p;
        float xr = src[0], xi = src[1];
        float* dst = out + (long)bs * (NH_ * QKR_) + h * QKR_ + 2 * p;
        dst[0] = xr * c - xi * sn;
        dst[1] = xr * sn + xi * c;
    }
}

__global__ void k_copy_kv(const float* __restrict__ kvf, float* __restrict__ out, int n) {
    int i = blockIdx.x * blockDim.x + threadIdx.x;
    if (i < n) {
        int d  = i & 511;
        int bs = i >> 9;
        out[i] = kvf[(long)bs * (KVL_ + QKR_) + d];
    }
}

__global__ void k_rot_kpe(const float* __restrict__ kvf,
                          const float* __restrict__ fc, const float* __restrict__ fs,
                          float* __restrict__ out, int n) {   // n = B*S*32 pairs
    int i = blockIdx.x * blockDim.x + threadIdx.x;
    if (i < n) {
        int p  = i & 31;
        int bs = i >> 5;
        int s  = bs & (S_ - 1);
        float c  = fc[s * 32 + p];
        float sn = fs[s * 32 + p];
        const float* src = kvf + (long)bs * (KVL_ + QKR_) + KVL_ + 2 * p;
        float xr = src[0], xi = src[1];
        float* dst = out + (long)bs * QKR_ + 2 * p;
        dst[0] = xr * c - xi * sn;
        dst[1] = xr * sn + xi * c;
    }
}

// ---------------------------------------------------------------------------
extern "C" void kernel_launch(void* const* d_in, const int* in_sizes, int n_in,
                              void* d_out, int out_size, void* d_ws, size_t ws_size,
                              hipStream_t stream) {
    const float* x     = (const float*)d_in[0];  // [B,S,DIM]
    const float* fc    = (const float*)d_in[1];  // [S,32]
    const float* fs    = (const float*)d_in[2];  // [S,32]
    const float* wq_a  = (const float*)d_in[3];  // [DIM,QL]
    const float* qnw   = (const float*)d_in[4];  // [QL]
    const float* wq_b  = (const float*)d_in[5];  // [QL, NH*QKH]
    const float* wkv_a = (const float*)d_in[6];  // [DIM, KVL+QKR]

    // ---- workspace layout (256B aligned chunks) ----
    char* ws = (char*)d_ws;
    size_t off = 0;
    auto alloc = [&](size_t bytes) {
        void* p = ws + off;
        off += (bytes + 255) & ~(size_t)255;
        return p;
    };
    unsigned short* xb      = (unsigned short*)alloc((size_t)MTOT * DIM_ * 2);              // [8192,2048] bf16
    unsigned short* wq_aT   = (unsigned short*)alloc((size_t)QL_ * DIM_ * 2);               // [1536,2048] bf16
    unsigned short* wq_bT   = (unsigned short*)alloc((size_t)(NH_ * QKH_) * QL_ * 2);       // [3072,1536] bf16
    unsigned short* wkv_aT  = (unsigned short*)alloc((size_t)(KVL_ + QKR_) * DIM_ * 2);     // [576,2048]  bf16
    float*          q_low   = (float*)alloc((size_t)MTOT * QL_ * 4);                        // [8192,1536] f32
    unsigned short* q_norm  = (unsigned short*)alloc((size_t)MTOT * QL_ * 2);               // [8192,1536] bf16
    float*          qfull   = (float*)alloc((size_t)MTOT * NH_ * QKH_ * 4);                 // [8192,3072] f32
    float*          kvfull  = (float*)alloc((size_t)MTOT * (KVL_ + QKR_) * 4);              // [8192,576]  f32
    (void)ws_size;

    // ---- output sub-buffers (flat concatenation, reference return order) ----
    float* out        = (float*)d_out;
    float* out_qnope  = out;                                        // B*S*16*128
    float* out_qpe    = out_qnope + (size_t)MTOT * NH_ * QKN_;      // B*S*16*64
    float* out_kv     = out_qpe   + (size_t)MTOT * NH_ * QKR_;      // B*S*512
    float* out_kpe    = out_kv    + (size_t)MTOT * KVL_;            // B*S*64

    const int T = 256;
    auto cdiv = [](int a, int b) { return (a + b - 1) / b; };

    // ---- stage 0: precision conversion (+ weight transpose for B frags) ----
    {
        int n = MTOT * DIM_;
        k_conv<<<cdiv(n, T), T, 0, stream>>>(x, xb, n);
    }
    k_conv_tr<<<cdiv(DIM_ * QL_, T), T, 0, stream>>>(wq_a, wq_aT, DIM_, QL_);
    k_conv_tr<<<cdiv(QL_ * NH_ * QKH_, T), T, 0, stream>>>(wq_b, wq_bT, QL_, NH_ * QKH_);
    k_conv_tr<<<cdiv(DIM_ * (KVL_ + QKR_), T), T, 0, stream>>>(wkv_a, wkv_aT, DIM_, KVL_ + QKR_);

    // ---- GEMM 1: q_low = x @ wq_a  (K=2048, N=1536) ----
    k_gemm_bf16<DIM_, QL_><<<dim3(cdiv(QL_, 256), MTOT / 128), T, 0, stream>>>(
        xb, wq_aT, q_low, MTOT);

    // ---- RMSNorm -> bf16 ----
    k_rmsnorm<<<MTOT, T, 0, stream>>>(q_low, qnw, q_norm);

    // ---- GEMM 2: q = q_norm @ wq_b  (K=1536, N=3072) ----
    k_gemm_bf16<QL_, NH_ * QKH_><<<dim3(cdiv(NH_ * QKH_, 256), MTOT / 128), T, 0, stream>>>(
        q_norm, wq_bT, qfull, MTOT);

    // ---- GEMM 3: kv_full = x @ wkv_a  (K=2048, N=576) ----
    k_gemm_bf16<DIM_, KVL_ + QKR_><<<dim3(cdiv(KVL_ + QKR_, 256), MTOT / 128), T, 0, stream>>>(
        xb, wkv_aT, kvfull, MTOT);

    // ---- epilogues: splits + rotary ----
    {
        int n = MTOT * NH_ * QKN_;
        k_copy_qnope<<<cdiv(n, T), T, 0, stream>>>(qfull, out_qnope, n);
    }
    {
        int n = MTOT * NH_ * 32;  // pairs
        k_rot_qpe<<<cdiv(n, T), T, 0, stream>>>(qfull, fc, fs, out_qpe, n);
    }
    {
        int n = MTOT * KVL_;
        k_copy_kv<<<cdiv(n, T), T, 0, stream>>>(kvfull, out_kv, n);
    }
    {
        int n = MTOT * 32;  // pairs
        k_rot_kpe<<<cdiv(n, T), T, 0, stream>>>(kvfull, fc, fs, out_kpe, n);
    }
}
